// match_block_65455301591739
// MI455X (gfx1250) — compile-verified, matrix operand order checked by reading, not run on hardware
//
#include <hip/hip_runtime.h>
#include <math.h>

typedef __attribute__((ext_vector_type(16))) _Float16 v16h;
typedef __attribute__((ext_vector_type(8)))  _Float16 v8h;
typedef __attribute__((ext_vector_type(8)))  float    v8f;

#define BM 128
#define BN 128
#define BKK 32
#define GEMM_THREADS 256

#define EPI_NONE 0   // D=f16: alpha*acc
#define EPI_BIAS 1   // D=f16: alpha*acc + bias[m]
#define EPI_FULL 2   // D=f32: BN(alpha*acc + bias[m]) + resid[m,n]

// D[M,N] = epilogue( alpha * op(A) @ op(B) )  with f16 inputs.
// TA=0: A row-major [M,K]; TA=1: A row-major [K,M] (use A^T)
// TB=0: B row-major [K,N]; TB=1: B row-major [N,K] (use B^T)
template<bool TA, bool TB, int EPI>
__global__ __launch_bounds__(GEMM_THREADS)
void wmma_gemm_kernel(const _Float16* __restrict__ A, long long sA,
                      const _Float16* __restrict__ Bm, long long sB,
                      void* __restrict__ Dv, long long sD,
                      const float* __restrict__ resid, long long sR,
                      const float* __restrict__ bias,
                      const float* __restrict__ bng,
                      const float* __restrict__ bnb,
                      int M, int N, int K, float alpha)
{
    // Both tiles fragment-major: row index = matrix row (A) / column (B), col = k.
    __shared__ _Float16 As[BM][BKK + 8];
    __shared__ _Float16 Bs[BN][BKK + 8];

    const int bz = blockIdx.z;
    A  += (long long)bz * sA;
    Bm += (long long)bz * sB;

    const int m0 = blockIdx.y * BM;
    const int n0 = blockIdx.x * BN;
    const int tid  = threadIdx.x;
    const int lane = tid & 31;
    const int wave = tid >> 5;
    const int wm = wave >> 1;   // 0..3 : 32-row strip
    const int wn = wave & 1;    // 0..1 : 64-col strip

    v8f acc[2][4] = {};

    const int h   = lane >> 4;   // lane half -> K sub-range
    const int l15 = lane & 15;
    const int h8  = h << 3;

    for (int k0 = 0; k0 < K; k0 += BKK) {
        // ---- stage A tile: As[m][k], 512 chunks of 8 halves, 2 per thread ----
        if constexpr (!TA) {
            #pragma unroll
            for (int it = 0; it < 2; ++it) {
                const int t8 = tid + it * GEMM_THREADS;
                const int r  = t8 >> 2;            // BKK/8 = 4 chunks per row
                const int c8 = (t8 & 3) << 3;
                const v8h v = *(const v8h*)(A + (long long)(m0 + r) * K + k0 + c8);
                *(v8h*)&As[r][c8] = v;
            }
        } else {
            #pragma unroll
            for (int it = 0; it < 2; ++it) {
                const int t8 = tid + it * GEMM_THREADS;
                const int kr = t8 >> 4;            // BM/8 = 16 chunks per k-row
                const int mc = (t8 & 15) << 3;
                const v8h v = *(const v8h*)(A + (long long)(k0 + kr) * M + m0 + mc);
                #pragma unroll
                for (int j = 0; j < 8; ++j) As[mc + j][kr] = v[j];
            }
        }
        // ---- stage B tile: Bs[n][k] ----
        if constexpr (!TB) {
            #pragma unroll
            for (int it = 0; it < 2; ++it) {
                const int t8 = tid + it * GEMM_THREADS;
                const int kr = t8 >> 4;            // BN/8 = 16 chunks per k-row
                const int nc = (t8 & 15) << 3;
                const v8h v = *(const v8h*)(Bm + (long long)(k0 + kr) * N + n0 + nc);
                #pragma unroll
                for (int j = 0; j < 8; ++j) Bs[nc + j][kr] = v[j];
            }
        } else {
            #pragma unroll
            for (int it = 0; it < 2; ++it) {
                const int t8  = tid + it * GEMM_THREADS;
                const int nr  = t8 >> 2;           // BKK/8 = 4 chunks per n-row
                const int kc8 = (t8 & 3) << 3;
                const v8h v = *(const v8h*)(Bm + (long long)(n0 + nr) * K + k0 + kc8);
                *(v8h*)&Bs[nr][kc8] = v;
            }
        }
        __syncthreads();

        // ---- fragments: each = two contiguous 16B chunks of one LDS row ----
        v16h bf[4];
        #pragma unroll
        for (int ns = 0; ns < 4; ++ns) {
            const _Float16* rp = &Bs[wn * 64 + ns * 16 + l15][0];
            const v8h b0 = *(const v8h*)(rp + h8);
            const v8h b1 = *(const v8h*)(rp + 16 + h8);
            bf[ns] = __builtin_shufflevector(b0, b1, 0, 1, 2, 3, 4, 5, 6, 7,
                                             8, 9, 10, 11, 12, 13, 14, 15);
        }
        #pragma unroll
        for (int ms = 0; ms < 2; ++ms) {
            const _Float16* rp = &As[wm * 32 + ms * 16 + l15][0];
            const v8h a0 = *(const v8h*)(rp + h8);
            const v8h a1 = *(const v8h*)(rp + 16 + h8);
            const v16h af = __builtin_shufflevector(a0, a1, 0, 1, 2, 3, 4, 5, 6, 7,
                                                    8, 9, 10, 11, 12, 13, 14, 15);
            #pragma unroll
            for (int ns = 0; ns < 4; ++ns) {
                acc[ms][ns] = __builtin_amdgcn_wmma_f32_16x16x32_f16(
                    false, af, false, bf[ns], (short)0, acc[ms][ns], false, false);
            }
        }
        __syncthreads();
    }

    // ---- epilogue (fully branch-free; per-fragment base pointer + stride walk) ----
    const int mhi = (lane >> 4) << 3;   // lanes 16-31 -> M += 8
    if constexpr (EPI != EPI_FULL) {
        _Float16* D = (_Float16*)Dv + (long long)bz * sD;
        #pragma unroll
        for (int ms = 0; ms < 2; ++ms) {
            const int mbase = m0 + wm * 32 + ms * 16 + mhi;
            float bv[8];
            #pragma unroll
            for (int r = 0; r < 8; ++r)
                bv[r] = (EPI == EPI_BIAS) ? bias[mbase + r] : 0.0f;
            #pragma unroll
            for (int ns = 0; ns < 4; ++ns) {
                const int nidx = n0 + wn * 64 + ns * 16 + l15;
                _Float16* p = D + (long long)mbase * N + nidx;
                #pragma unroll
                for (int r = 0; r < 8; ++r) {
                    *p = (_Float16)(acc[ms][ns][r] * alpha + bv[r]);
                    p += N;
                }
            }
        }
    } else {
        const float rinv = rsqrtf(1.0f + 1e-5f);
        float* D = (float*)Dv + (long long)bz * sD;
        const float* Rp = resid + (long long)bz * sR;
        #pragma unroll
        for (int ms = 0; ms < 2; ++ms) {
            const int mbase = m0 + wm * 32 + ms * 16 + mhi;
            float bv[8], sc[8], sh[8];
            #pragma unroll
            for (int r = 0; r < 8; ++r) {
                bv[r] = bias[mbase + r];
                sc[r] = bng[mbase + r] * rinv;
                sh[r] = bnb[mbase + r];
            }
            #pragma unroll
            for (int ns = 0; ns < 4; ++ns) {
                const int nidx = n0 + wn * 64 + ns * 16 + l15;
                const long long base = (long long)mbase * N + nidx;
                float* p = D + base;
                const float* q = Rp + base;
                #pragma unroll
                for (int r = 0; r < 8; ++r) {
                    *p = (acc[ms][ns][r] + bv[r]) * sc[r] + sh[r] + *q;
                    p += N;
                    q += N;
                }
            }
        }
    }
}

__global__ void f32_to_f16_kernel(const float* __restrict__ in,
                                  _Float16* __restrict__ out, long long n)
{
    const long long i = ((long long)blockIdx.x * blockDim.x + threadIdx.x) * 4;
    if (i >= n) return;
    const float4 v = *(const float4*)(in + i);
    out[i + 0] = (_Float16)v.x;
    out[i + 1] = (_Float16)v.y;
    out[i + 2] = (_Float16)v.z;
    out[i + 3] = (_Float16)v.w;
}

// avg + max over the spatial dim of one [C,Nsp] row per block (Nsp == blockDim.x == 256)
__global__ void pool_avgmax_kernel(const float* __restrict__ x,
                                   float* __restrict__ avg, float* __restrict__ mx, int n)
{
    __shared__ float ss[256];
    __shared__ float sm[256];
    const int row = blockIdx.x;
    const int tid = threadIdx.x;
    const float v = x[(long long)row * n + tid];
    ss[tid] = v; sm[tid] = v;
    __syncthreads();
    for (int s = 128; s > 0; s >>= 1) {
        if (tid < s) {
            ss[tid] += ss[tid + s];
            sm[tid] = fmaxf(sm[tid], sm[tid + s]);
        }
        __syncthreads();
    }
    if (tid == 0) { avg[row] = ss[0] / (float)n; mx[row] = sm[0]; }
}

__global__ void gate_mlp1_kernel(const float* __restrict__ avg, const float* __restrict__ mx,
                                 const float* __restrict__ w1, const float* __restrict__ b1,
                                 float* __restrict__ ha, float* __restrict__ hm,
                                 int Bn, int C, int R)
{
    const int idx = blockIdx.x * blockDim.x + threadIdx.x;
    if (idx >= Bn * R) return;
    const int b = idx >> 6, r = idx & 63;          // R == 64
    const float* wr = w1 + (long long)r * C;
    const float* pa = avg + (long long)b * C;
    const float* pm = mx + (long long)b * C;
    float sa = 0.f, sm = 0.f;
    for (int c = 0; c < C; ++c) { const float w = wr[c]; sa += w * pa[c]; sm += w * pm[c]; }
    ha[idx] = fmaxf(sa + b1[r], 0.f);
    hm[idx] = fmaxf(sm + b1[r], 0.f);
}

__global__ void gate_mlp2_kernel(const float* __restrict__ ha, const float* __restrict__ hm,
                                 const float* __restrict__ w2, const float* __restrict__ b2,
                                 float* __restrict__ cw, int Bn, int C, int R)
{
    const int idx = blockIdx.x * blockDim.x + threadIdx.x;
    if (idx >= Bn * C) return;
    const int b = idx >> 10, c = idx & 1023;       // C == 1024
    const float* wc = w2 + (long long)c * R;
    const float* pa = ha + (long long)b * R;
    const float* pm = hm + (long long)b * R;
    float sa = 0.f, sm = 0.f;
    for (int r = 0; r < R; ++r) { const float w = wc[r]; sa += w * pa[r]; sm += w * pm[r]; }
    const float v = (sa + b2[c]) + (sm + b2[c]);
    cw[idx] = 1.0f / (1.0f + expf(-v));
}

// grid: (Nsp/256, B*C); no integer division anywhere
__global__ void gate_scale_kernel(const float* __restrict__ x, const float* __restrict__ cw,
                                  float* __restrict__ out, int Nsp)
{
    const long long bc = blockIdx.y;
    const int n = blockIdx.x * 256 + threadIdx.x;
    const long long i = bc * Nsp + n;
    out[i] = x[i] * cw[bc];
}

extern "C" void kernel_launch(void* const* d_in, const int* in_sizes, int n_in,
                              void* d_out, int out_size, void* d_ws, size_t ws_size,
                              hipStream_t stream)
{
    (void)in_sizes; (void)n_in; (void)out_size; (void)ws_size;

    const int Bn = 8, C = 1024, Ci = 512;
    const int Nd = 48 * 48;   // 2304
    const int Na = 16 * 16;   // 256
    const int R  = 64;

    const float* detect  = (const float*)d_in[0];
    const float* aim     = (const float*)d_in[1];
    const float* g_w     = (const float*)d_in[2];
    const float* g_b     = (const float*)d_in[3];
    const float* theta_w = (const float*)d_in[4];
    const float* theta_b = (const float*)d_in[5];
    const float* phi_w   = (const float*)d_in[6];
    const float* phi_b   = (const float*)d_in[7];
    const float* W_w     = (const float*)d_in[8];
    const float* W_b     = (const float*)d_in[9];
    const float* W_bn_g  = (const float*)d_in[10];
    const float* W_bn_b  = (const float*)d_in[11];
    const float* Q_w     = (const float*)d_in[12];
    const float* Q_b     = (const float*)d_in[13];
    const float* Q_bn_g  = (const float*)d_in[14];
    const float* Q_bn_b  = (const float*)d_in[15];
    const float* mlp1_w  = (const float*)d_in[16];
    const float* mlp1_b  = (const float*)d_in[17];
    const float* mlp2_w  = (const float*)d_in[18];
    const float* mlp2_b  = (const float*)d_in[19];

    // ---- workspace carve-out (f16 region then f32 region) ----
    char* base = (char*)d_ws;
    auto take16 = [&](long long nhalf) {
        _Float16* p = (_Float16*)base; base += nhalf * 2; return p;
    };
    _Float16* detect16 = take16((long long)Bn * C * Nd);
    _Float16* aim16    = take16((long long)Bn * C * Na);
    _Float16* gw16     = take16((long long)Ci * C);
    _Float16* thw16    = take16((long long)Ci * C);
    _Float16* phw16    = take16((long long)Ci * C);
    _Float16* Ww16     = take16((long long)C * Ci);
    _Float16* Qw16     = take16((long long)C * Ci);
    _Float16* dx16     = take16((long long)Bn * Ci * Nd);  // later non_det_cm
    _Float16* phi16    = take16((long long)Bn * Ci * Nd);
    _Float16* ax16     = take16((long long)Bn * Ci * Na);
    _Float16* theta16  = take16((long long)Bn * Ci * Na);
    _Float16* fm16     = take16((long long)Bn * Na * Nd);
    _Float16* naim16   = take16((long long)Bn * Ci * Na);
    float* naimf = (float*)base; base += (long long)Bn * C * Na * 4;
    float* pavg  = (float*)base; base += (long long)Bn * C * 4;
    float* pmax  = (float*)base; base += (long long)Bn * C * 4;
    float* ha    = (float*)base; base += (long long)Bn * R * 4;
    float* hm    = (float*)base; base += (long long)Bn * R * 4;

    float* out_nondet = (float*)d_out;
    float* out_actdet = out_nondet + (long long)Bn * C * Nd;
    float* out_actaim = out_actdet + (long long)Bn * C * Nd;
    float* out_cw     = out_actaim + (long long)Bn * C * Na;

    // ---- one-time f32 -> f16 conversion of all GEMM operands ----
    auto cvt = [&](const float* src, _Float16* dst, long long n) {
        f32_to_f16_kernel<<<(unsigned)((n / 4 + 255) / 256), 256, 0, stream>>>(src, dst, n);
    };
    cvt(detect,  detect16, (long long)Bn * C * Nd);
    cvt(aim,     aim16,    (long long)Bn * C * Na);
    cvt(g_w,     gw16,     (long long)Ci * C);
    cvt(theta_w, thw16,    (long long)Ci * C);
    cvt(phi_w,   phw16,    (long long)Ci * C);
    cvt(W_w,     Ww16,     (long long)C * Ci);
    cvt(Q_w,     Qw16,     (long long)C * Ci);

    const dim3 blk(GEMM_THREADS);
    const long long sDet = (long long)C * Nd, sAim = (long long)C * Na;

    // 1) d_x_cm [Ci,Nd] = g_w @ detect + g_b        (f16 out)
    wmma_gemm_kernel<false, false, EPI_BIAS><<<dim3(Nd / BN, Ci / BM, Bn), blk, 0, stream>>>(
        gw16, 0, detect16, sDet, dx16, (long long)Ci * Nd,
        nullptr, 0, g_b, nullptr, nullptr, Ci, Nd, C, 1.0f);
    // 2) a_x_cm [Ci,Na] = g_w @ aim + g_b
    wmma_gemm_kernel<false, false, EPI_BIAS><<<dim3(Na / BN, Ci / BM, Bn), blk, 0, stream>>>(
        gw16, 0, aim16, sAim, ax16, (long long)Ci * Na,
        nullptr, 0, g_b, nullptr, nullptr, Ci, Na, C, 1.0f);
    // 3) theta_cm [Ci,Na] = theta_w @ aim + theta_b
    wmma_gemm_kernel<false, false, EPI_BIAS><<<dim3(Na / BN, Ci / BM, Bn), blk, 0, stream>>>(
        thw16, 0, aim16, sAim, theta16, (long long)Ci * Na,
        nullptr, 0, theta_b, nullptr, nullptr, Ci, Na, C, 1.0f);
    // 4) phi_cm [Ci,Nd] = phi_w @ detect + phi_b
    wmma_gemm_kernel<false, false, EPI_BIAS><<<dim3(Nd / BN, Ci / BM, Bn), blk, 0, stream>>>(
        phw16, 0, detect16, sDet, phi16, (long long)Ci * Nd,
        nullptr, 0, phi_b, nullptr, nullptr, Ci, Nd, C, 1.0f);
    // 5) f [Na,Nd] = theta_cm^T @ phi_cm            (TN)
    wmma_gemm_kernel<true, false, EPI_NONE><<<dim3(Nd / BN, Na / BM, Bn), blk, 0, stream>>>(
        theta16, (long long)Ci * Na, phi16, (long long)Ci * Nd, fm16, (long long)Na * Nd,
        nullptr, 0, nullptr, nullptr, nullptr, Na, Nd, Ci, 1.0f);
    // 6) non_aim_cm [Ci,Na] = (1/Nd) * d_x_cm @ f^T (NT)
    wmma_gemm_kernel<false, true, EPI_NONE><<<dim3(Na / BN, Ci / BM, Bn), blk, 0, stream>>>(
        dx16, (long long)Ci * Nd, fm16, (long long)Na * Nd, naim16, (long long)Ci * Na,
        nullptr, 0, nullptr, nullptr, nullptr, Ci, Na, Nd, 1.0f / (float)Nd);
    // 7) non_det_cm [Ci,Nd] = (1/Na) * a_x_cm @ f   (NN, reuses dx16 buffer)
    wmma_gemm_kernel<false, false, EPI_NONE><<<dim3(Nd / BN, Ci / BM, Bn), blk, 0, stream>>>(
        ax16, (long long)Ci * Na, fm16, (long long)Na * Nd, dx16, (long long)Ci * Nd,
        nullptr, 0, nullptr, nullptr, nullptr, Ci, Nd, Na, 1.0f / (float)Na);
    // 8) non_aim final (f32) = BN(W_w @ non_aim_cm + W_b) + aim
    wmma_gemm_kernel<false, false, EPI_FULL><<<dim3(Na / BN, C / BM, Bn), blk, 0, stream>>>(
        Ww16, 0, naim16, (long long)Ci * Na, naimf, sAim,
        aim, sAim, W_b, W_bn_g, W_bn_b, C, Na, Ci, 1.0f);
    // 9) non_det final (f32) = BN(Q_w @ non_det_cm + Q_b) + detect -> d_out[0]
    wmma_gemm_kernel<false, false, EPI_FULL><<<dim3(Nd / BN, C / BM, Bn), blk, 0, stream>>>(
        Qw16, 0, dx16, (long long)Ci * Nd, out_nondet, sDet,
        detect, sDet, Q_b, Q_bn_g, Q_bn_b, C, Nd, Ci, 1.0f);

    // ---- channel gate on non_aim final ----
    pool_avgmax_kernel<<<Bn * C, 256, 0, stream>>>(naimf, pavg, pmax, Na);
    gate_mlp1_kernel<<<(Bn * R + 255) / 256, 256, 0, stream>>>(
        pavg, pmax, mlp1_w, mlp1_b, ha, hm, Bn, C, R);
    gate_mlp2_kernel<<<(Bn * C + 255) / 256, 256, 0, stream>>>(
        ha, hm, mlp2_w, mlp2_b, out_cw, Bn, C, R);

    // ---- gated outputs ----
    gate_scale_kernel<<<dim3(Nd / 256, Bn * C), 256, 0, stream>>>(
        out_nondet, out_cw, out_actdet, Nd);
    gate_scale_kernel<<<dim3(Na / 256, Bn * C), 256, 0, stream>>>(
        naimf, out_cw, out_actaim, Na);
}
